// StructBiasPooler_69707319214602
// MI455X (gfx1250) — compile-verified, matrix operand order checked by reading, not run on hardware
//
#include <hip/hip_runtime.h>
#include <hip/hip_bf16.h>
#include <cstddef>
#include <cstdint>

// ---------------- problem constants ----------------
#define BB     4
#define LLEN   2048
#define DM     640
#define DS     64
#define KKDIM  128
#define NQKV   192      // q|k|v concatenated along N
#define NCOMP  10
#define WINHALF 16      // band: m-l in [-16, 15]
#define BANDW  48       // union of band for 16 consecutive rows

typedef __bf16 bf16_t;
typedef __attribute__((ext_vector_type(16))) __bf16 v16bf;
typedef __attribute__((ext_vector_type(8)))  float  v8f;
typedef __attribute__((ext_vector_type(4)))  unsigned int v4u;
typedef __attribute__((ext_vector_type(8)))  int v8i;
typedef __attribute__((ext_vector_type(4)))  int v4i;

#if defined(__AMDGCN__) && __has_builtin(__builtin_amdgcn_tensor_load_to_lds) && \
    __has_builtin(__builtin_amdgcn_s_wait_tensorcnt)
#define USE_TDM 1
#else
#define USE_TDM 0
#endif

// ---------------- WMMA fragment helpers (ISA 7.12.2, wave32) ----------------
// 16-bit A 16x32: lanes 0-15 hold row M=lane, K = {0..7, 16..23};
// lanes 16-31 hold row M=lane-16, K = {8..15, 24..31}. VGPR v packs 2 bf16.
__device__ __forceinline__ int frag_koff(int v, int h) {
    return ((v & 4) ? 16 : 0) + h * 8 + (v & 3) * 2;
}

// Row-of-a-row-major-matrix fragment (works for A tiles, and for B tiles when
// the B matrix is stored TRANSPOSED row-major: then "row0" = output column n0).
// Per lane this is two contiguous 16-byte runs -> b128 loads (global or ds).
__device__ __forceinline__ v16bf load_row_frag(const bf16_t* A, int lda,
                                               int row0, int k0) {
    int lane = threadIdx.x & 31;
    int r = row0 + (lane & 15);
    int h = lane >> 4;
    v16bf f;
#pragma unroll
    for (int v = 0; v < 8; ++v) {
        int k = k0 + frag_koff(v, h);
        f[2 * v]     = A[(size_t)r * lda + k];
        f[2 * v + 1] = A[(size_t)r * lda + k + 1];
    }
    return f;
}

#if USE_TDM
// ---------------- Tensor Data Mover: 2-D bf16 tile -> LDS (ISA ch.8 D#) -----
__device__ __forceinline__ void tdm_load_2d_bf16(const bf16_t* gsrc, uint32_t lds_off,
                                                 uint32_t width, uint32_t rows,
                                                 uint32_t row_stride, uint32_t tensor_rows) {
    uint64_t ga = (uint64_t)(uintptr_t)gsrc;
    v4u g0;
    g0[0] = 1u;                                   // count=1, user descriptor
    g0[1] = lds_off;                              // lds_addr (bytes)
    g0[2] = (uint32_t)(ga & 0xffffffffu);         // global_addr[31:0]
    g0[3] = (uint32_t)((ga >> 32) & 0x01ffffffu)  // global_addr[56:32]
            | 0x80000000u;                        // type=2 ("image")
    v8i g1;
    g1[0] = (int)(1u << 16);                                          // data_size=1 (2B), wg_mask=0
    g1[1] = (int)((width & 0xffffu) << 16);                           // tensor_dim0 lo16
    g1[2] = (int)((width >> 16) | ((tensor_rows & 0xffffu) << 16));   // dim0 hi | tensor_dim1 lo
    g1[3] = (int)((tensor_rows >> 16) | ((width & 0xffffu) << 16));   // dim1 hi | tile_dim0
    g1[4] = (int)(rows & 0xffffu);                                    // tile_dim1 | tile_dim2=0
    g1[5] = (int)row_stride;                                          // tensor_dim0_stride lo32
    g1[6] = 0;
    g1[7] = 0;
    v4i gz = {0, 0, 0, 0};
#if __clang_major__ >= 23
    v8i gz8 = {0, 0, 0, 0, 0, 0, 0, 0};
    __builtin_amdgcn_tensor_load_to_lds(g0, g1, gz, gz, gz8, 0);
#else
    __builtin_amdgcn_tensor_load_to_lds(g0, g1, gz, gz, 0);
#endif
}
#endif

// ---------------- small prep kernels ----------------
__global__ void k_cast_bf16(const float* __restrict__ src, bf16_t* __restrict__ dst, int n) {
    int i = blockIdx.x * blockDim.x + threadIdx.x;
    int stride = gridDim.x * blockDim.x;
    for (; i < n; i += stride) dst[i] = (bf16_t)src[i];
}

// WqkvT[n][k] : n in [0,192) selects q/k/v column, k in [0,640)
__global__ void k_pack_wqkvT(const float* __restrict__ Wq, const float* __restrict__ Wk,
                             const float* __restrict__ Wv, bf16_t* __restrict__ out) {
    int i = blockIdx.x * blockDim.x + threadIdx.x;
    if (i >= NQKV * DM) return;
    int n = i / DM, k = i % DM;
    float v;
    if (n < 64)       v = Wq[k * 64 + n];
    else if (n < 128) v = Wk[k * 64 + (n - 64)];
    else              v = Wv[k * 64 + (n - 128)];
    out[i] = (bf16_t)v;
}

// WcvT[n][k] : n in [0,128), k in [0,64)
__global__ void k_pack_wcvT(const float* __restrict__ Wcv, bf16_t* __restrict__ out) {
    int i = blockIdx.x * blockDim.x + threadIdx.x;
    if (i >= KKDIM * DS) return;
    int n = i / DS, k = i % DS;
    out[i] = (bf16_t)Wcv[k * KKDIM + n];
}

// struct MLP: feat[5] -> relu(128) -> g[64]; also ||g||^2
__global__ void k_struct_mlp(const int* __restrict__ ss3, const float* __restrict__ rsa,
                             const float* __restrict__ plddt,
                             const float* __restrict__ W1, const float* __restrict__ b1,
                             const float* __restrict__ W2, const float* __restrict__ b2,
                             float* __restrict__ g, float* __restrict__ sq) {
    int i = blockIdx.x * blockDim.x + threadIdx.x;
    if (i >= BB * LLEN) return;
    int s3 = ss3[i];
    float fr = rsa[i], fp = plddt[i];
    float acc[DS];
#pragma unroll
    for (int d = 0; d < DS; ++d) acc[d] = b2[d];
    for (int j = 0; j < 128; ++j) {
        float h = W1[s3 * 128 + j] + fr * W1[3 * 128 + j] + fp * W1[4 * 128 + j] + b1[j];
        h = h > 0.f ? h : 0.f;
#pragma unroll
        for (int d = 0; d < DS; ++d) acc[d] += h * W2[j * DS + d];
    }
    float s = 0.f;
#pragma unroll
    for (int d = 0; d < DS; ++d) {
        g[(size_t)i * DS + d] = acc[d];
        s += acc[d] * acc[d];
    }
    sq[i] = s;
}

// ---------------- GEMM1: qkv = esm_bf16[8192x640] x Wqkv[640x192] ----------------
// grid (512, 3), block 128 (4 waves, one 16x16 N-tile each, shared A tile).
// A tile (16x640, 20KB) staged into LDS once per block via the Tensor Data Mover;
// B comes from the pre-transposed weights (contiguous per-lane b128 loads).
__global__ void __launch_bounds__(128) k_gemm_qkv(
    const bf16_t* __restrict__ Abf, const bf16_t* __restrict__ WqkvT,
    const float* __restrict__ bq, const float* __restrict__ bk, const float* __restrict__ bv,
    bf16_t* __restrict__ qs, bf16_t* __restrict__ kb, bf16_t* __restrict__ vb) {
    __shared__ bf16_t smA[16 * DM];               // 20 KB
    int wave = threadIdx.x >> 5;
    int lane = threadIdx.x & 31;
    int m0 = blockIdx.x * 16;
    int n0 = (blockIdx.y * 4 + wave) * 16;

#if USE_TDM
    if (threadIdx.x < 32) {                       // wave 0 issues the TDM DMA
        tdm_load_2d_bf16(Abf + (size_t)m0 * DM,
                         (uint32_t)(uintptr_t)&smA[0],
                         /*width=*/DM, /*rows=*/16,
                         /*row_stride=*/DM, /*tensor_rows=*/BB * LLEN);
        __builtin_amdgcn_s_wait_tensorcnt(0);
    }
    asm volatile("" ::: "memory");                // LDS written behind compiler's back
#else
    for (int i = threadIdx.x; i < 16 * DM; i += 128)
        smA[i] = Abf[(size_t)m0 * DM + i];
#endif
    __syncthreads();

    v8f acc = {};
#pragma unroll 4
    for (int kt = 0; kt < DM / 32; ++kt) {
        int k0 = kt * 32;
        if (kt + 1 < DM / 32)                     // gfx1250 global_prefetch_b8 path
            __builtin_prefetch(&WqkvT[(size_t)(n0 + (lane & 15)) * DM + k0 + 32], 0, 1);
        v16bf a = load_row_frag(smA, DM, 0, k0);          // ds_load_b128 x2
        v16bf b = load_row_frag(WqkvT, DM, n0, k0);       // global_load_b128 x2
        acc = __builtin_amdgcn_wmma_f32_16x16x32_bf16(false, a, false, b,
                                                      (short)0, acc, false, false);
    }
    int h = lane >> 4;
    int col = n0 + (lane & 15);
    float bias = (col < 64) ? bq[col] : (col < 128) ? bk[col - 64] : bv[col - 128];
#pragma unroll
    for (int r = 0; r < 8; ++r) {
        int m = m0 + r + 8 * h;
        float val = acc[r] + bias;
        if (col < 64)        qs[(size_t)m * DS + col]         = (bf16_t)(val * 0.125f); // fold 1/sqrt(64)
        else if (col < 128)  kb[(size_t)m * DS + (col - 64)]  = (bf16_t)val;
        else                 vb[(size_t)m * DS + (col - 128)] = (bf16_t)val;
    }
}

// ---------------- GEMM2: v_ = vb[8192x64] x Wcv[64x128] + bcv ----------------
// grid (512, 2), block 128; B from transposed WcvT
__global__ void __launch_bounds__(128) k_gemm_vcv(
    const bf16_t* __restrict__ vb, const bf16_t* __restrict__ WcvT,
    const float* __restrict__ bcv, float* __restrict__ vproj) {
    int wave = threadIdx.x >> 5;
    int lane = threadIdx.x & 31;
    int m0 = blockIdx.x * 16;
    int n0 = (blockIdx.y * 4 + wave) * 16;
    v8f acc = {};
#pragma unroll
    for (int kt = 0; kt < 2; ++kt) {
        v16bf a = load_row_frag(vb, DS, m0, kt * 32);
        v16bf b = load_row_frag(WcvT, DS, n0, kt * 32);
        acc = __builtin_amdgcn_wmma_f32_16x16x32_bf16(false, a, false, b,
                                                      (short)0, acc, false, false);
    }
    int h = lane >> 4;
    int col = n0 + (lane & 15);
#pragma unroll
    for (int r = 0; r < 8; ++r)
        vproj[(size_t)(m0 + r + 8 * h) * KKDIM + col] = acc[r] + bcv[col];
}

// ---------------- per-batch pooling weights: w = softmax(plddt/5) ----------------
__global__ void k_softmax_w(const float* __restrict__ plddt, float* __restrict__ w) {
    int b = blockIdx.x, tid = threadIdx.x;
    __shared__ float red[256];
    float mx = -1e30f;
    for (int l = tid; l < LLEN; l += 256) mx = fmaxf(mx, plddt[b * LLEN + l] * 0.2f);
    red[tid] = mx; __syncthreads();
    for (int s = 128; s > 0; s >>= 1) { if (tid < s) red[tid] = fmaxf(red[tid], red[tid + s]); __syncthreads(); }
    float m = red[0]; __syncthreads();
    float sum = 0.f;
    for (int l = tid; l < LLEN; l += 256) sum += __expf(plddt[b * LLEN + l] * 0.2f - m);
    red[tid] = sum; __syncthreads();
    for (int s = 128; s > 0; s >>= 1) { if (tid < s) red[tid] += red[tid + s]; __syncthreads(); }
    float inv = 1.f / red[0]; __syncthreads();
    for (int l = tid; l < LLEN; l += 256) w[b * LLEN + l] = __expf(plddt[b * LLEN + l] * 0.2f - m) * inv;
}

// ---------------- banded attention: one block per (b, 16-row tile) ----------------
// wave 0: 16x48 logit strip via 3 WMMA N-tiles (K-loop of 2) -> LDS
// all threads: +bias/mask, row softmax, banded context
__global__ void __launch_bounds__(256) k_attn(
    const bf16_t* __restrict__ qs, const bf16_t* __restrict__ kb,
    const float* __restrict__ g, const float* __restrict__ sq,
    const float* __restrict__ vproj,
    const float* __restrict__ lamp, const float* __restrict__ betap,
    float* __restrict__ ctx) {
    int b  = blockIdx.y;
    int l0 = blockIdx.x * 16;
    int tid = threadIdx.x;

    __shared__ float sL[16][BANDW];
    __shared__ float sA[16][BANDW];

    const bf16_t* qsb = qs + (size_t)b * LLEN * DS;
    const bf16_t* kbb = kb + (size_t)b * LLEN * DS;

    if (tid < 32) {              // whole wave 0, EXEC all ones
        int lane = tid;
        int h = lane >> 4;
#pragma unroll
        for (int t = 0; t < 3; ++t) {
            v8f acc = {};
            int mbase = l0 - WINHALF + 16 * t;
#pragma unroll
            for (int kt = 0; kt < 2; ++kt) {
                int k0 = kt * 32;
                v16bf a = load_row_frag(qsb, DS, l0, k0);
                v16bf bb;                       // B(k,n) = kbb[mbase+n][k]  (contiguous per lane)
                {
                    int n = lane & 15;
                    int m = mbase + n;
                    m = m < 0 ? 0 : (m > LLEN - 1 ? LLEN - 1 : m);
#pragma unroll
                    for (int v = 0; v < 8; ++v) {
                        int k = k0 + frag_koff(v, h);
                        bb[2 * v]     = kbb[(size_t)m * DS + k];
                        bb[2 * v + 1] = kbb[(size_t)m * DS + k + 1];
                    }
                }
                acc = __builtin_amdgcn_wmma_f32_16x16x32_bf16(false, a, false, bb,
                                                              (short)0, acc, false, false);
            }
#pragma unroll
            for (int r = 0; r < 8; ++r)
                sL[r + 8 * h][16 * t + (lane & 15)] = acc[r];
        }
    }
    __syncthreads();

    float lam = *lamp, beta = *betap;
    // mask + structural bias (fp32), in place: one thread owns each element
    for (int e = tid; e < 16 * BANDW; e += 256) {
        int row = e / BANDW, c = e % BANDW;
        int l = l0 + row;
        int m = l0 - WINHALF + c;
        float val;
        if (m < 0 || m >= LLEN) {
            val = -1e30f;                       // column doesn't exist in reference
        } else {
            int dlt = c - WINHALF - row;        // == m - l
            float logit = (dlt >= -16 && dlt <= 15) ? sL[row][c] : -1e9f;
            const float* gl = g + ((size_t)b * LLEN + l) * DS;
            const float* gm = g + ((size_t)b * LLEN + m) * DS;
            float gg = 0.f;
#pragma unroll
            for (int d = 0; d < DS; ++d) gg += gl[d] * gm[d];
            float d2 = sq[b * LLEN + l] + sq[b * LLEN + m] - 2.f * gg;
            float dist = sqrtf(fmaxf(d2, 0.f));
            val = logit + lam * gg - beta * dist;
        }
        sL[row][c] = val;
    }
    __syncthreads();

    if (tid < 16) {                              // row softmax over 48 cols
        int row = tid;
        float mx = -1e30f;
        for (int c = 0; c < BANDW; ++c) mx = fmaxf(mx, sL[row][c]);
        float sum = 0.f;
        for (int c = 0; c < BANDW; ++c) { float ex = __expf(sL[row][c] - mx); sA[row][c] = ex; sum += ex; }
        float inv = 1.f / sum;
        for (int c = 0; c < BANDW; ++c) sA[row][c] *= inv;
    }
    __syncthreads();

    // banded context: 16 rows x 128 k = 2048 outputs, 8 per thread
#pragma unroll
    for (int o = 0; o < 8; ++o) {
        int idx = tid + 256 * o;
        int row = idx >> 7;
        int kk  = idx & 127;
        float acc = 0.f;
        for (int c = 0; c < BANDW; ++c) {
            int m = l0 - WINHALF + c;
            int mc = m < 0 ? 0 : (m >= LLEN ? LLEN - 1 : m);   // attn==0 there
            acc += sA[row][c] * vproj[((size_t)b * LLEN + mc) * KKDIM + kk];
        }
        ctx[((size_t)b * LLEN + l0 + row) * KKDIM + kk] = acc;
    }
}

// ---------------- pool + head: z = tanh(sum_l w*ctx); out = sigmoid(z Wh + bh) ----
__global__ void __launch_bounds__(128) k_pool_head(
    const float* __restrict__ w, const float* __restrict__ ctx,
    const float* __restrict__ Wh, const float* __restrict__ bh, float* __restrict__ out) {
    int b = blockIdx.x, k = threadIdx.x;          // 128 threads
    float acc = 0.f;
    for (int l = 0; l < LLEN; ++l)                // deterministic serial reduction
        acc += w[b * LLEN + l] * ctx[((size_t)b * LLEN + l) * KKDIM + k];
    __shared__ float z[KKDIM];
    z[k] = tanhf(acc);
    __syncthreads();
    if (k < NCOMP) {
        float s = bh[k];
        for (int kk = 0; kk < KKDIM; ++kk) s += z[kk] * Wh[kk * NCOMP + k];
        out[b * NCOMP + k] = 1.f / (1.f + __expf(-s));
    }
}

// ---------------- host launcher ----------------
extern "C" void kernel_launch(void* const* d_in, const int* in_sizes, int n_in,
                              void* d_out, int out_size, void* d_ws, size_t ws_size,
                              hipStream_t stream) {
    const float* esm   = (const float*)d_in[0];
    const float* plddt = (const float*)d_in[1];
    const int*   ss3   = (const int*)  d_in[2];
    const float* rsa   = (const float*)d_in[3];
    const float* W1    = (const float*)d_in[4];
    const float* b1    = (const float*)d_in[5];
    const float* W2    = (const float*)d_in[6];
    const float* b2    = (const float*)d_in[7];
    const float* Wq    = (const float*)d_in[8];
    const float* bq    = (const float*)d_in[9];
    const float* Wk    = (const float*)d_in[10];
    const float* bk    = (const float*)d_in[11];
    const float* Wv    = (const float*)d_in[12];
    const float* bv    = (const float*)d_in[13];
    const float* lam   = (const float*)d_in[14];
    const float* beta  = (const float*)d_in[15];
    const float* Wcv   = (const float*)d_in[16];
    const float* bcv   = (const float*)d_in[17];
    const float* Wh    = (const float*)d_in[18];
    const float* bh    = (const float*)d_in[19];
    float* out = (float*)d_out;

    // workspace carve-out (256B aligned), total ~24.5 MB
    size_t off = 0;
    char* base = (char*)d_ws;
    auto take = [&](size_t bytes) -> char* {
        char* r = base + off;
        off += (bytes + 255) & ~(size_t)255;
        return r;
    };
    const int MROWS = BB * LLEN;                              // 8192
    bf16_t* esm_bf  = (bf16_t*)take((size_t)MROWS * DM * sizeof(bf16_t));
    bf16_t* wqkvT   = (bf16_t*)take((size_t)NQKV * DM * sizeof(bf16_t));
    bf16_t* wcvT    = (bf16_t*)take((size_t)KKDIM * DS * sizeof(bf16_t));
    bf16_t* qs_bf   = (bf16_t*)take((size_t)MROWS * DS * sizeof(bf16_t));
    bf16_t* kb_bf   = (bf16_t*)take((size_t)MROWS * DS * sizeof(bf16_t));
    bf16_t* vb_bf   = (bf16_t*)take((size_t)MROWS * DS * sizeof(bf16_t));
    float*  g_ws    = (float*) take((size_t)MROWS * DS * sizeof(float));
    float*  sq_ws   = (float*) take((size_t)MROWS * sizeof(float));
    float*  vproj   = (float*) take((size_t)MROWS * KKDIM * sizeof(float));
    float*  ctx_ws  = (float*) take((size_t)MROWS * KKDIM * sizeof(float));
    float*  w_ws    = (float*) take((size_t)MROWS * sizeof(float));
    (void)in_sizes; (void)n_in; (void)out_size; (void)ws_size;

    // 1) precision conversion / weight packing (transposed for WMMA B operand)
    k_cast_bf16<<<512, 256, 0, stream>>>(esm, esm_bf, MROWS * DM);
    k_pack_wqkvT<<<(NQKV * DM + 255) / 256, 256, 0, stream>>>(Wq, Wk, Wv, wqkvT);
    k_pack_wcvT<<<(KKDIM * DS + 255) / 256, 256, 0, stream>>>(Wcv, wcvT);

    // 2) struct MLP -> g, ||g||^2
    k_struct_mlp<<<MROWS / 256, 256, 0, stream>>>(ss3, rsa, plddt, W1, b1, W2, b2,
                                                  g_ws, sq_ws);

    // 3) fused QKV projection (WMMA bf16, TDM-staged A tile)
    k_gemm_qkv<<<dim3(MROWS / 16, 3), 128, 0, stream>>>(esm_bf, wqkvT, bq, bk, bv,
                                                        qs_bf, kb_bf, vb_bf);

    // 4) v_ = v @ Wcv + bcv (WMMA bf16)
    k_gemm_vcv<<<dim3(MROWS / 16, 2), 128, 0, stream>>>(vb_bf, wcvT, bcv, vproj);

    // 5) pooling weights
    k_softmax_w<<<BB, 256, 0, stream>>>(plddt, w_ws);

    // 6) banded attention (WMMA logits + fp32 bias/softmax/context)
    k_attn<<<dim3(LLEN / 16, BB), 256, 0, stream>>>(qs_bf, kb_bf, g_ws, sq_ws, vproj,
                                                    lam, beta, ctx_ws);

    // 7) pool + head
    k_pool_head<<<BB, 128, 0, stream>>>(w_ws, ctx_ws, Wh, bh, out);
}
